// MultiHeadSelfAttention_16054587752878
// MI455X (gfx1250) — compile-verified, matrix operand order checked by reading, not run on hardware
//
#include <hip/hip_runtime.h>
#include <stdint.h>

// Problem constants (match reference)
constexpr int BB = 4, SS = 2048, EE = 1024, HH = 16, DD = 64;
constexpr int MM = BB * SS; // 8192 token rows

typedef __bf16 bf16;
typedef __attribute__((ext_vector_type(16))) __bf16 v16bf;
typedef __attribute__((ext_vector_type(8)))  __bf16 v8bf;
typedef __attribute__((ext_vector_type(4)))  __bf16 v4bf;
typedef __attribute__((ext_vector_type(8)))  float  v8f;
typedef __attribute__((ext_vector_type(4)))  int    v4i;

// ---- optional CDNA5 async global->LDS path (ASYNCcnt), gated on toolchain
#if defined(__has_builtin)
#  if __has_builtin(__builtin_amdgcn_global_load_async_to_lds_b128) && \
      __has_builtin(__builtin_amdgcn_s_wait_asynccnt)
#    define HAVE_ASYNC_LDS 1
#  endif
#endif
#ifndef HAVE_ASYNC_LDS
#  define HAVE_ASYNC_LDS 0
#endif

static __device__ __forceinline__ void cp16_g2l(const bf16* g, bf16* l) {
#if HAVE_ASYNC_LDS
  // Signature (from clang diagnostic): (v4i*, v4i*, int offset, int cpol)
  __builtin_amdgcn_global_load_async_to_lds_b128((v4i*)g, (v4i*)l, 0, 0);
#else
  *(v8bf*)l = *(const v8bf*)g;
#endif
}
static __device__ __forceinline__ void async_wait0() {
#if HAVE_ASYNC_LDS
  __builtin_amdgcn_s_wait_asynccnt(0);
#endif
}

// Build a 16x32 bf16 WMMA operand fragment from LDS.
// Per CDNA5 ISA layout (16-bit A 16x32): lane holds row (lane&15); the 16
// halfwords are K = [8*hi .. 8*hi+7] then [16+8*hi .. 16+8*hi+7] (hi=lane>>4).
// Caller passes p = &row[8*hi]; second chunk is p+16 elements.
static __device__ __forceinline__ v16bf ldfrag(const bf16* p) {
  v8bf lo = *(const v8bf*)(p);
  v8bf hi = *(const v8bf*)(p + 16);
  v16bf r;
#pragma unroll
  for (int i = 0; i < 8; i++) { r[i] = lo[i]; r[i + 8] = hi[i]; }
  return r;
}

#define WMMA_BF16(a, b, c) \
  __builtin_amdgcn_wmma_f32_16x16x32_bf16(false, (a), false, (b), (short)0, (c), false, false)

static __device__ __forceinline__ v8f v8f_zero() {
  v8f z = {0.f, 0.f, 0.f, 0.f, 0.f, 0.f, 0.f, 0.f};
  return z;
}

// ---------------------------------------------------------------- converts
__global__ void cvt_f32_bf16_x4(const float4* __restrict__ s,
                                v4bf* __restrict__ d, int n4) {
  int i = blockIdx.x * blockDim.x + threadIdx.x;
  int st = gridDim.x * blockDim.x;
  for (; i < n4; i += st) {
    float4 f = s[i];
    v4bf o;
    o[0] = (bf16)f.x; o[1] = (bf16)f.y; o[2] = (bf16)f.z; o[3] = (bf16)f.w;
    d[i] = o;
  }
}

__global__ void mask_prep(const uint8_t* __restrict__ m,
                          float* __restrict__ mb, int n) {
  int i = blockIdx.x * blockDim.x + threadIdx.x;
  if (i < n) mb[i] = m[i] ? 0.0f : -1.0e6f;
}

// ---------------------------------------------------------------- GEMM
// C[m,n] = sum_k A[m,k] * W[n,k] + bias[n]   (i.e. y = A @ W^T + b)
// Block tile 128x256xK64; 8 waves as 2(M) x 4(N); wave tile 64x64.
// Per 32-K step a wave does 8 fragment loads for 16 WMMAs.
// EPI==0: store bf16 into Cb. EPI==1: store fp32 abs(v)*mask into Cf.
constexpr int GBM = 128, GBN = 256, GBK = 64;
constexpr int LDK = 72; // padded bf16 row -> conflict-free b128 frag reads

template <int EPI>
__global__ __launch_bounds__(256) void gemm_bf16_wmma(
    const bf16* __restrict__ A, const bf16* __restrict__ W,
    const float* __restrict__ bias, bf16* __restrict__ Cb,
    float* __restrict__ Cf, const uint8_t* __restrict__ mask,
    int Mdim, int Ndim, int Kdim) {
  __shared__ __align__(16) bf16 As[GBM * LDK]; // 18 KB
  __shared__ __align__(16) bf16 Bs[GBN * LDK]; // 36 KB

  const int tid = threadIdx.x;
  const int lane = tid & 31, wave = tid >> 5;
  const int hl = lane & 15, hi = lane >> 4;
  const int nTiles = Ndim / GBN;
  const int m0 = (blockIdx.x / nTiles) * GBM;
  const int n0 = (blockIdx.x % nTiles) * GBN;
  const int waveM = (wave & 1) * 64;  // 2 waves along M
  const int waveN = (wave >> 1) * 64; // 4 waves along N

  v8f acc[4][4];
#pragma unroll
  for (int i = 0; i < 4; i++)
#pragma unroll
    for (int j = 0; j < 4; j++) acc[i][j] = v8f_zero();

  for (int kt = 0; kt < Kdim; kt += GBK) {
    // Stage A (128x64) and B (256x64) tiles; 16B chunks per lane.
#pragma unroll
    for (int c = tid; c < (GBM * GBK) / 8; c += 256) { // 4 per thread
      int row = c >> 3, kc = (c & 7) * 8;
      cp16_g2l(&A[(size_t)(m0 + row) * Kdim + kt + kc], &As[row * LDK + kc]);
    }
#pragma unroll
    for (int c = tid; c < (GBN * GBK) / 8; c += 256) { // 8 per thread
      int row = c >> 3, kc = (c & 7) * 8;
      cp16_g2l(&W[(size_t)(n0 + row) * Kdim + kt + kc], &Bs[row * LDK + kc]);
    }
    async_wait0();
    __syncthreads();

#pragma unroll
    for (int ks = 0; ks < 2; ks++) {
      const int k0 = ks * 32 + 8 * hi;
      v16bf a4[4];
#pragma unroll
      for (int tm = 0; tm < 4; tm++)
        a4[tm] = ldfrag(&As[(waveM + tm * 16 + hl) * LDK + k0]);
#pragma unroll
      for (int tn = 0; tn < 4; tn++) {
        v16bf bw = ldfrag(&Bs[(waveN + tn * 16 + hl) * LDK + k0]);
#pragma unroll
        for (int tm = 0; tm < 4; tm++)
          acc[tm][tn] = WMMA_BF16(a4[tm], bw, acc[tm][tn]);
      }
    }
    __syncthreads();
  }

  // Epilogue. C layout: lane hl holds N=tile_n+hl; VGPR r holds M = r + 8*hi.
#pragma unroll
  for (int tm = 0; tm < 4; tm++) {
#pragma unroll
    for (int tn = 0; tn < 4; tn++) {
      const int n = n0 + waveN + tn * 16 + hl;
      const int mrow = m0 + waveM + tm * 16 + 8 * hi;
      const float bv = bias[n];
#pragma unroll
      for (int r = 0; r < 8; r++) {
        const int m = mrow + r;
        float v = acc[tm][tn][r] + bv;
        if (EPI == 0) {
          Cb[(size_t)m * Ndim + n] = (bf16)v;
        } else {
          const int bidx = m >> 11;          // m / SS
          const int s = m & (SS - 1);        // m % SS
          v = mask[bidx * SS + s] ? fabsf(v) : 0.0f;
          Cf[(size_t)m * Ndim + n] = v;
        }
      }
    }
  }
}

// ---------------------------------------------------------------- attention
// Flash attention, bf16 WMMA, fp32 online softmax. One block = 128 queries of
// one (b,h); 8 waves x 16 queries. Key blocks of 64.
constexpr int AQ = 128, KBLK = 64, LDH = 72;

__global__ __launch_bounds__(256) void attn_fwd_wmma(
    const bf16* __restrict__ Q, const bf16* __restrict__ K,
    const bf16* __restrict__ V, const float* __restrict__ maskbias,
    bf16* __restrict__ O) {
  __shared__ __align__(16) bf16 Qs[AQ * LDH];
  __shared__ __align__(16) bf16 Ks[KBLK * LDH];
  __shared__ __align__(16) bf16 Vt[DD * LDH];        // transposed: [d][key]
  __shared__ __align__(16) bf16 Pb[8 * 16 * LDH];    // per-wave P (16 x 64)

  const int tid = threadIdx.x;
  const int lane = tid & 31, wave = tid >> 5;
  const int hl = lane & 15, hi = lane >> 4;
  const int qt = blockIdx.x & 15;         // S/AQ = 16 query tiles
  const int h = (blockIdx.x >> 4) & 15;
  const int b = blockIdx.x >> 8;

  const bf16* Qg = Q + ((size_t)b * SS + (size_t)qt * AQ) * EE + h * DD;
  // Stage the 128x64 Q tile once (1024 chunks of 8 bf16, 4 per thread).
#pragma unroll
  for (int c = tid; c < 1024; c += 256) {
    int row = c >> 3, kc = (c & 7) * 8;
    cp16_g2l(&Qg[(size_t)row * EE + kc], &Qs[row * LDH + kc]);
  }
  async_wait0();

  v8f accO[4];
#pragma unroll
  for (int tn = 0; tn < 4; tn++) accO[tn] = v8f_zero();
  float mrun[8], lrun[8];
#pragma unroll
  for (int r = 0; r < 8; r++) { mrun[r] = -1.0e30f; lrun[r] = 0.0f; }

  bf16* Pw = &Pb[wave * 16 * LDH];

  for (int kb = 0; kb < SS; kb += KBLK) {
    __syncthreads(); // Q staged (iter 0) / previous AV reads done
    const bf16* Kg = K + ((size_t)b * SS + kb) * EE + h * DD;
    const bf16* Vg = V + ((size_t)b * SS + kb) * EE + h * DD;
    // Stage 64x64 K (row-major, async) and V transposed ([d][key]).
#pragma unroll
    for (int c = tid; c < 512; c += 256) {
      int row = c >> 3, kc = (c & 7) * 8;
      cp16_g2l(&Kg[(size_t)row * EE + kc], &Ks[row * LDH + kc]);
      v8bf vv = *(const v8bf*)&Vg[(size_t)row * EE + kc];
#pragma unroll
      for (int j = 0; j < 8; j++) Vt[(kc + j) * LDH + row] = vv[j];
    }
    async_wait0();
    __syncthreads();

    // S = Q K^T : M=16 queries, N=64 keys, K=D=64 (2 wmma k-steps).
    v8f accS[4];
#pragma unroll
    for (int tn = 0; tn < 4; tn++) accS[tn] = v8f_zero();
#pragma unroll
    for (int ks = 0; ks < 2; ks++) {
      const int k0 = ks * 32 + 8 * hi;
      v16bf aq = ldfrag(&Qs[(wave * 16 + hl) * LDH + k0]);
#pragma unroll
      for (int tn = 0; tn < 4; tn++) {
        v16bf bk = ldfrag(&Ks[(tn * 16 + hl) * LDH + k0]);
        accS[tn] = WMMA_BF16(aq, bk, accS[tn]);
      }
    }

    // Online softmax, all in registers. Row m = r + 8*hi lives in VGPR r,
    // its 16 N-values are striped across the 16 lanes of this half-wave.
    float rmax[8];
#pragma unroll
    for (int r = 0; r < 8; r++) rmax[r] = -1.0e30f;
#pragma unroll
    for (int tn = 0; tn < 4; tn++) {
      const float mb = maskbias[(size_t)b * SS + kb + tn * 16 + hl];
#pragma unroll
      for (int r = 0; r < 8; r++) {
        float sv = accS[tn][r] * 0.125f + mb; // 1/sqrt(D) = 0.125
        accS[tn][r] = sv;
        rmax[r] = fmaxf(rmax[r], sv);
      }
    }
#pragma unroll
    for (int mk = 1; mk < 16; mk <<= 1)
#pragma unroll
      for (int r = 0; r < 8; r++)
        rmax[r] = fmaxf(rmax[r], __shfl_xor(rmax[r], mk, 32));

    float alpha[8], rsum[8];
#pragma unroll
    for (int r = 0; r < 8; r++) {
      float nm = fmaxf(mrun[r], rmax[r]);
      alpha[r] = __expf(mrun[r] - nm);
      mrun[r] = nm;
      rsum[r] = 0.0f;
    }
#pragma unroll
    for (int tn = 0; tn < 4; tn++)
#pragma unroll
      for (int r = 0; r < 8; r++) {
        float e = __expf(accS[tn][r] - mrun[r]);
        accS[tn][r] = e;
        rsum[r] += e;
      }
#pragma unroll
    for (int mk = 1; mk < 16; mk <<= 1)
#pragma unroll
      for (int r = 0; r < 8; r++) rsum[r] += __shfl_xor(rsum[r], mk, 32);
#pragma unroll
    for (int r = 0; r < 8; r++) lrun[r] = lrun[r] * alpha[r] + rsum[r];

    // Reshape P (C-layout) -> A-layout via LDS; rescale running O.
#pragma unroll
    for (int tn = 0; tn < 4; tn++)
#pragma unroll
      for (int r = 0; r < 8; r++) {
        Pw[(r + 8 * hi) * LDH + tn * 16 + hl] = (bf16)accS[tn][r];
        accO[tn][r] *= alpha[r];
      }
    __syncthreads();

    // O += P V : M=16 queries, N=64 d, K=64 keys (2 wmma k-steps).
#pragma unroll
    for (int ks = 0; ks < 2; ks++) {
      const int k0 = ks * 32 + 8 * hi;
      v16bf ap = ldfrag(&Pw[hl * LDH + k0]);
#pragma unroll
      for (int tn = 0; tn < 4; tn++) {
        v16bf bv = ldfrag(&Vt[(tn * 16 + hl) * LDH + k0]);
        accO[tn] = WMMA_BF16(ap, bv, accO[tn]);
      }
    }
  }

  // Normalize by 1/l and store bf16 O in (B,S,E) layout.
#pragma unroll
  for (int tn = 0; tn < 4; tn++) {
    const int d = tn * 16 + hl;
#pragma unroll
    for (int r = 0; r < 8; r++) {
      const int q = qt * AQ + wave * 16 + r + 8 * hi;
      O[((size_t)b * SS + q) * EE + h * DD + d] =
          (bf16)(accO[tn][r] * (1.0f / lrun[r]));
    }
  }
}

// ---------------------------------------------------------------- launch
extern "C" void kernel_launch(void* const* d_in, const int* in_sizes, int n_in,
                              void* d_out, int out_size, void* d_ws,
                              size_t ws_size, hipStream_t stream) {
  const float* x = (const float*)d_in[0];
  const uint8_t* mask = (const uint8_t*)d_in[1]; // numpy bool_: 1 byte
  const float* WQw = (const float*)d_in[2];
  const float* WQb = (const float*)d_in[3];
  const float* WKw = (const float*)d_in[4];
  const float* WKb = (const float*)d_in[5];
  const float* WVw = (const float*)d_in[6];
  const float* WVb = (const float*)d_in[7];
  const float* WOw = (const float*)d_in[8];
  const float* WOb = (const float*)d_in[9];
  float* out = (float*)d_out;

  // Workspace carve-up (~88 MiB total)
  char* ws = (char*)d_ws;
  size_t off = 0;
  auto take = [&](size_t bytes) { char* p = ws + off; off += bytes; return p; };
  bf16* xb = (bf16*)take((size_t)MM * EE * 2);
  bf16* wq = (bf16*)take((size_t)EE * EE * 2);
  bf16* wk = (bf16*)take((size_t)EE * EE * 2);
  bf16* wv = (bf16*)take((size_t)EE * EE * 2);
  bf16* wo = (bf16*)take((size_t)EE * EE * 2);
  bf16* qb = (bf16*)take((size_t)MM * EE * 2);
  bf16* kb = (bf16*)take((size_t)MM * EE * 2);
  bf16* vb = (bf16*)take((size_t)MM * EE * 2);
  bf16* ob = (bf16*)take((size_t)MM * EE * 2);
  float* mbias = (float*)take((size_t)BB * SS * 4);
  (void)ws_size; (void)in_sizes; (void)n_in; (void)out_size;

  // fp32 -> bf16 converts
  {
    int n4 = MM * EE / 4;
    cvt_f32_bf16_x4<<<2048, 256, 0, stream>>>((const float4*)x, (v4bf*)xb, n4);
    int w4 = EE * EE / 4;
    cvt_f32_bf16_x4<<<512, 256, 0, stream>>>((const float4*)WQw, (v4bf*)wq, w4);
    cvt_f32_bf16_x4<<<512, 256, 0, stream>>>((const float4*)WKw, (v4bf*)wk, w4);
    cvt_f32_bf16_x4<<<512, 256, 0, stream>>>((const float4*)WVw, (v4bf*)wv, w4);
    cvt_f32_bf16_x4<<<512, 256, 0, stream>>>((const float4*)WOw, (v4bf*)wo, w4);
    mask_prep<<<(BB * SS + 255) / 256, 256, 0, stream>>>(mask, mbias, BB * SS);
  }

  // Q/K/V projections: 8192x1024x1024 bf16 WMMA GEMMs
  const int gemmBlocks = (MM / GBM) * (EE / GBN); // 64 * 4 = 256
  gemm_bf16_wmma<0><<<gemmBlocks, 256, 0, stream>>>(xb, wq, WQb, qb, nullptr,
                                                    nullptr, MM, EE, EE);
  gemm_bf16_wmma<0><<<gemmBlocks, 256, 0, stream>>>(xb, wk, WKb, kb, nullptr,
                                                    nullptr, MM, EE, EE);
  gemm_bf16_wmma<0><<<gemmBlocks, 256, 0, stream>>>(xb, wv, WVb, vb, nullptr,
                                                    nullptr, MM, EE, EE);

  // Flash attention: B*H*(S/128) = 1024 blocks
  attn_fwd_wmma<<<BB * HH * (SS / AQ), 256, 0, stream>>>(qb, kb, vb, mbias, ob);

  // Output projection with fused abs(mask * .) epilogue -> fp32 d_out
  gemm_bf16_wmma<1><<<gemmBlocks, 256, 0, stream>>>(ob, wo, WOb, nullptr, out,
                                                    mask, MM, EE, EE);
}
